// DualModel_53266184405810
// MI455X (gfx1250) — matrix operand order repaired and checked by reading, not compiled, not run on hardware
//
#include <hip/hip_runtime.h>
#include <stdint.h>

#define D_MODEL 256

typedef __attribute__((ext_vector_type(16))) __bf16 v16bf;
typedef __attribute__((ext_vector_type(8)))  float  v8f;
typedef __attribute__((ext_vector_type(8)))  unsigned short u16x8;
typedef __attribute__((ext_vector_type(4)))  unsigned int   u32x4;
typedef __attribute__((ext_vector_type(8)))  int            i32x8;
typedef __attribute__((ext_vector_type(4)))  int            i32x4;

#if defined(__HIP_DEVICE_COMPILE__) && defined(__gfx1250__) && \
    __has_builtin(__builtin_amdgcn_tensor_load_to_lds)
#define HAVE_TDM 1
#else
#define HAVE_TDM 0
#endif

union Frag {
  v16bf h;
  u16x8 u[2];
};

__device__ __forceinline__ unsigned short f2bf(float f) {
  union { float f; unsigned int u; } v; v.f = f;
  unsigned int r = v.u + 0x7FFFu + ((v.u >> 16) & 1u);
  return (unsigned short)(r >> 16);
}

__device__ __forceinline__ v8f zero8() {
  v8f z = {0.f, 0.f, 0.f, 0.f, 0.f, 0.f, 0.f, 0.f};
  return z;
}

__device__ __forceinline__ v8f wmma_bf16(const Frag& a, const Frag& b, v8f c) {
  return __builtin_amdgcn_wmma_f32_16x16x32_bf16(false, a.h, false, b.h,
                                                 (short)0, c, false, false);
}

// ---------------------------------------------------------------------------
// X = E[input] + P ; also produce bf16 copy
// grid: (B*S) blocks, 256 threads
// ---------------------------------------------------------------------------
__global__ void embed_kernel(const int* __restrict__ inp,
                             const float* __restrict__ E,
                             const float* __restrict__ P,
                             float* __restrict__ Xf,
                             unsigned short* __restrict__ Xb,
                             int S) {
  const int bs = blockIdx.x;
  const int d  = threadIdx.x;
  const int tok = inp[bs];
  const float x = E[(size_t)tok * D_MODEL + d] + P[(size_t)(bs % S) * D_MODEL + d];
  Xf[(size_t)bs * D_MODEL + d] = x;
  Xb[(size_t)bs * D_MODEL + d] = f2bf(x);
}

// fp32 -> bf16 convert
__global__ void cvt_kernel(const float* __restrict__ in,
                           unsigned short* __restrict__ out, int n) {
  int i = blockIdx.x * blockDim.x + threadIdx.x;
  if (i < n) out[i] = f2bf(in[i]);
}

// ---------------------------------------------------------------------------
// C[m,n] = scale * sum_k A[m,k] * W[n,k]  (+ optional residual)
// A: (M x 256) bf16 row-major, W: (N x 256) bf16 row-major (i.e. computes A @ W^T)
// One wave per 16x16 tile; 128-thread block covers 16(M) x 64(N).
// grid: (N/64, M/16)
// ---------------------------------------------------------------------------
__global__ void __launch_bounds__(128)
gemm_bf16_kernel(const unsigned short* __restrict__ A,
                 const unsigned short* __restrict__ W,
                 const float* __restrict__ resid,
                 float* __restrict__ outF,
                 unsigned short* __restrict__ outB,
                 int N, float scale) {
  const int wave = threadIdx.x >> 5;
  const int lane = threadIdx.x & 31;
  const int row  = lane & 15;
  const int half = lane >> 4;
  const int m0 = blockIdx.y << 4;
  const int n0 = (blockIdx.x << 6) + (wave << 4);

  const unsigned short* Ar = A + (size_t)(m0 + row) * D_MODEL;
  const unsigned short* Wr = W + (size_t)(n0 + row) * D_MODEL;

  v8f acc = zero8();
#pragma unroll
  for (int kb = 0; kb < D_MODEL; kb += 32) {
    Frag a, b;
    // A fragment: lane = (row m, half); element e -> K = 16*half + e
    a.u[0] = *(const u16x8*)(Ar + kb + half * 16);
    a.u[1] = *(const u16x8*)(Ar + kb + half * 16 + 8);
    // B fragment: lane = (col n, half); element e -> K = 16*half + e
    b.u[0] = *(const u16x8*)(Wr + kb + half * 16);
    b.u[1] = *(const u16x8*)(Wr + kb + half * 16 + 8);
    if (kb + 32 < D_MODEL) {
      __builtin_prefetch(Ar + kb + 32, 0, 3);
      __builtin_prefetch(Wr + kb + 32, 0, 3);
    }
    acc = wmma_bf16(a, b, acc);
  }

  const int n = n0 + row;
#pragma unroll
  for (int r = 0; r < 8; ++r) {
    const int m = m0 + r + half * 8;       // C layout: VGPR r, half -> row
    const size_t idx = (size_t)m * N + n;
    float v = acc[r] * scale;
    if (resid) v += resid[idx];
    if (outF) outF[idx] = v;
    if (outB) outB[idx] = f2bf(v);
  }
}

// ---------------------------------------------------------------------------
// Flash attention (causal), one wave = 16 query rows, block = 4 waves = 64 rows.
// Q pre-scaled by sqrt(D). Streams keys in 32-wide tiles with online softmax.
// K tile staged to LDS by the Tensor Data Mover (TENSOR_LOAD_TO_LDS +
// s_wait_tensorcnt); V tile staged transposed for contiguous B-fragments.
// grid: (S/64, B), block 128.
// ---------------------------------------------------------------------------
__global__ void __launch_bounds__(128)
attn_kernel(const unsigned short* __restrict__ Qb,
            const unsigned short* __restrict__ Kb,
            const unsigned short* __restrict__ Vb,
            const float* __restrict__ Xin,
            float* __restrict__ Xout,
            unsigned short* __restrict__ Xoutb,
            int S) {
  __shared__ __align__(16) unsigned short Kl[32 * D_MODEL];   // [k][d] row-major
  __shared__ __align__(16) unsigned short Vl[D_MODEL * 32];   // [d][k] transposed
  __shared__ __align__(16) unsigned short Pl[4][16 * 32];     // per-wave [m][k]

  const int tid  = threadIdx.x;
  const int wave = tid >> 5;
  const int lane = tid & 31;
  const int row  = lane & 15;
  const int half = lane >> 4;
  const int bb   = blockIdx.y;
  const int q0   = blockIdx.x << 6;
  const int m0   = q0 + (wave << 4);

  const unsigned short* Qr = Qb + ((size_t)(bb * S) + m0 + row) * D_MODEL;

  v8f o[16];
#pragma unroll
  for (int i = 0; i < 16; ++i) o[i] = zero8();
  float rm[8], rl[8];
#pragma unroll
  for (int r = 0; r < 8; ++r) { rm[r] = -3.0e38f; rl[r] = 0.f; }

  const int tmax = q0 + 64;   // causal: keys < q0+64 suffice for all rows in block
  for (int t0 = 0; t0 < tmax; t0 += 32) {
    __syncthreads();

    // ---- stage K tile (32 x 256 bf16, no transpose) via TDM ----
#if HAVE_TDM
    if (wave == 0) {
      const unsigned long long ga =
          (unsigned long long)(uintptr_t)(Kb + ((size_t)(bb * S) + t0) * D_MODEL);
      const unsigned lds = (unsigned)(uintptr_t)(&Kl[0]);
      u32x4 g0;
      g0.x = 1u;                                   // count=1 valid user descriptor
      g0.y = lds;                                  // D#.lds_addr
      g0.z = (unsigned)(ga & 0xFFFFFFFFu);         // D#.global_addr[31:0]
      g0.w = (unsigned)((ga >> 32) & 0x01FFFFFFu)  // D#.global_addr[56:32]
             | (2u << 30);                         // D#.type = 2 ("image")
      i32x8 g1;
      g1[0] = (int)(1u << 16);                     // data_size = 2 bytes
      g1[1] = (int)(256u << 16);                   // tensor_dim0[15:0] = 256
      g1[2] = (int)(((unsigned)S & 0xFFFFu) << 16); // tensor_dim1[15:0] = S
      g1[3] = (int)((((unsigned)S >> 16) & 0xFFFFu) | (256u << 16)); // tile_dim0=256
      g1[4] = 32;                                  // tile_dim1 = 32, tile_dim2 = 0
      g1[5] = 256;                                 // tensor_dim0_stride = 256
      g1[6] = 0;
      g1[7] = 0;
      i32x4 z4 = {0, 0, 0, 0};
#if __clang_major__ >= 23
      i32x8 z8 = {0, 0, 0, 0, 0, 0, 0, 0};
      __builtin_amdgcn_tensor_load_to_lds(g0, g1, z4, z4, z8, 0);
#else
      __builtin_amdgcn_tensor_load_to_lds(g0, g1, z4, z4, 0);
#endif
      __builtin_amdgcn_s_wait_tensorcnt(0);
    }
#else
    {
      // fallback: coalesced flat copy (rows are contiguous)
      const u16x8* src = (const u16x8*)(Kb + ((size_t)(bb * S) + t0) * D_MODEL);
      u16x8* dst = (u16x8*)Kl;
      for (int i = tid; i < (32 * D_MODEL) / 8; i += 128) dst[i] = src[i];
    }
#endif

    // ---- stage V tile transposed into LDS: Vl[d][k] = V[t0+k][d] ----
    for (int idx = tid; idx < 32 * D_MODEL; idx += 128) {
      const int k = idx >> 8;       // 0..31
      const int d = idx & 255;
      Vl[d * 32 + k] = Vb[((size_t)(bb * S) + t0 + k) * D_MODEL + d];
    }
    __syncthreads();

    // ---- scores: 16x32 tile as two 16x16 WMMA accumulators (K from LDS) ----
    v8f c0 = zero8(), c1 = zero8();
    const unsigned short* K0l = Kl + (size_t)row * D_MODEL;
    const unsigned short* K1l = Kl + (size_t)(16 + row) * D_MODEL;
#pragma unroll
    for (int kb = 0; kb < D_MODEL; kb += 32) {
      Frag a, b0, b1;
      a.u[0]  = *(const u16x8*)(Qr + kb + half * 16);
      a.u[1]  = *(const u16x8*)(Qr + kb + half * 16 + 8);
      b0.u[0] = *(const u16x8*)(K0l + kb + half * 16);
      b0.u[1] = *(const u16x8*)(K0l + kb + half * 16 + 8);
      b1.u[0] = *(const u16x8*)(K1l + kb + half * 16);
      b1.u[1] = *(const u16x8*)(K1l + kb + half * 16 + 8);
      c0 = wmma_bf16(a, b0, c0);
      c1 = wmma_bf16(a, b1, c1);
    }

    // ---- causal mask + online softmax (wave32: reduce within 16-lane half) --
#pragma unroll
    for (int r = 0; r < 8; ++r) {
      const int m   = m0 + r + half * 8;
      const int ng0 = t0 + row;
      float s0 = (ng0 <= m)      ? c0[r] : -1.0e30f;
      float s1 = (ng0 + 16 <= m) ? c1[r] : -1.0e30f;
      float tm = fmaxf(s0, s1);
      tm = fmaxf(tm, __shfl_xor(tm, 1, 32));
      tm = fmaxf(tm, __shfl_xor(tm, 2, 32));
      tm = fmaxf(tm, __shfl_xor(tm, 4, 32));
      tm = fmaxf(tm, __shfl_xor(tm, 8, 32));
      const float nm   = fmaxf(rm[r], tm);
      const float corr = __expf(rm[r] - nm);
      const float p0 = __expf(s0 - nm);
      const float p1 = __expf(s1 - nm);
      float ps = p0 + p1;
      ps += __shfl_xor(ps, 1, 32);
      ps += __shfl_xor(ps, 2, 32);
      ps += __shfl_xor(ps, 4, 32);
      ps += __shfl_xor(ps, 8, 32);
      rl[r] = rl[r] * corr + ps;
      rm[r] = nm;
      // stash P (bf16) in LDS in [m][k] order for A-fragment reload
      Pl[wave][(r + half * 8) * 32 + row]      = f2bf(p0);
      Pl[wave][(r + half * 8) * 32 + 16 + row] = f2bf(p1);
#pragma unroll
      for (int nt = 0; nt < 16; ++nt) o[nt][r] *= corr;
    }

    // wave-local LDS RAW: DS ops are in-order per wave; wait + compiler fence
    asm volatile("s_wait_dscnt 0" ::: "memory");

    // ---- O += P @ V : A-frag from Pl, B-frags from transposed Vl ----
    Frag p;
    p.u[0] = *(const u16x8*)&Pl[wave][row * 32 + half * 16];
    p.u[1] = *(const u16x8*)&Pl[wave][row * 32 + half * 16 + 8];
#pragma unroll
    for (int nt = 0; nt < 16; ++nt) {
      Frag vf;
      const unsigned short* vp = &Vl[(nt * 16 + row) * 32 + half * 16];
      vf.u[0] = *(const u16x8*)vp;
      vf.u[1] = *(const u16x8*)(vp + 8);
      o[nt] = wmma_bf16(p, vf, o[nt]);
    }
  }

  // ---- epilogue: normalize, residual, store fp32 + bf16 ----
#pragma unroll
  for (int r = 0; r < 8; ++r) {
    const float inv = 1.0f / rl[r];
    const int m = m0 + r + half * 8;
    const size_t base = ((size_t)(bb * S) + m) * D_MODEL;
#pragma unroll
    for (int nt = 0; nt < 16; ++nt) {
      const size_t idx = base + nt * 16 + row;
      const float v = o[nt][r] * inv + Xin[idx];
      Xout[idx]  = v;
      Xoutb[idx] = f2bf(v);
    }
  }
}

// ---------------------------------------------------------------------------
extern "C" void kernel_launch(void* const* d_in, const int* in_sizes, int n_in,
                              void* d_out, int out_size, void* d_ws, size_t ws_size,
                              hipStream_t stream) {
  const int*   inp  = (const int*)  d_in[0];
  // d_in[1] = causal mask (unused; mask applied analytically)
  const float* E    = (const float*)d_in[2];
  const float* P    = (const float*)d_in[3];
  const float* WQK1 = (const float*)d_in[4];
  const float* WOV1 = (const float*)d_in[5];
  const float* WQK2 = (const float*)d_in[6];
  const float* WOV2 = (const float*)d_in[7];
  const float* WF   = (const float*)d_in[8];
  const float* U    = (const float*)d_in[9];

  const int BS = in_sizes[0];              // B*S
  const int S  = in_sizes[3] / D_MODEL;    // P is (S, D)
  const int B  = BS / S;
  const int V  = in_sizes[9] / D_MODEL;    // U is (V, D)
  const int M  = BS;

  // bump allocator on workspace
  char* ws = (char*)d_ws;
  auto alloc = [&](size_t bytes) -> char* {
    char* p = ws;
    ws += (bytes + 255) & ~(size_t)255;
    return p;
  };
  float*          XfA  = (float*)         alloc((size_t)M * D_MODEL * sizeof(float));
  float*          XfB  = (float*)         alloc((size_t)M * D_MODEL * sizeof(float));
  unsigned short* XbA  = (unsigned short*)alloc((size_t)M * D_MODEL * 2);
  unsigned short* XbB  = (unsigned short*)alloc((size_t)M * D_MODEL * 2);
  unsigned short* Qbuf = (unsigned short*)alloc((size_t)M * D_MODEL * 2);
  unsigned short* Vbuf = (unsigned short*)alloc((size_t)M * D_MODEL * 2);
  unsigned short* wqk1b = (unsigned short*)alloc((size_t)D_MODEL * D_MODEL * 2);
  unsigned short* wov1b = (unsigned short*)alloc((size_t)D_MODEL * D_MODEL * 2);
  unsigned short* wqk2b = (unsigned short*)alloc((size_t)D_MODEL * D_MODEL * 2);
  unsigned short* wov2b = (unsigned short*)alloc((size_t)D_MODEL * D_MODEL * 2);
  unsigned short* wfb   = (unsigned short*)alloc((size_t)D_MODEL * D_MODEL * 2);
  unsigned short* ub    = (unsigned short*)alloc((size_t)V * D_MODEL * 2);

  const float sqd = 16.0f;  // sqrt(D_MODEL) folded into Q

  // weight conversions
  const int nw = D_MODEL * D_MODEL;
  cvt_kernel<<<(nw + 255) / 256, 256, 0, stream>>>(WQK1, wqk1b, nw);
  cvt_kernel<<<(nw + 255) / 256, 256, 0, stream>>>(WOV1, wov1b, nw);
  cvt_kernel<<<(nw + 255) / 256, 256, 0, stream>>>(WQK2, wqk2b, nw);
  cvt_kernel<<<(nw + 255) / 256, 256, 0, stream>>>(WOV2, wov2b, nw);
  cvt_kernel<<<(nw + 255) / 256, 256, 0, stream>>>(WF,   wfb,   nw);
  const int nu = V * D_MODEL;
  cvt_kernel<<<(nu + 255) / 256, 256, 0, stream>>>(U, ub, nu);

  // embedding
  embed_kernel<<<M, D_MODEL, 0, stream>>>(inp, E, P, XfA, XbA, S);

  dim3 gblk(128);
  dim3 gD(D_MODEL / 64, M / 16);          // D x D projections
  dim3 ablk(128);
  dim3 agrid(S / 64, B);

  // layer 1
  gemm_bf16_kernel<<<gD, gblk, 0, stream>>>(XbA, wqk1b, nullptr, nullptr, Qbuf, D_MODEL, sqd);
  gemm_bf16_kernel<<<gD, gblk, 0, stream>>>(XbA, wov1b, nullptr, nullptr, Vbuf, D_MODEL, 1.0f);
  attn_kernel<<<agrid, ablk, 0, stream>>>(Qbuf, XbA, Vbuf, XfA, XfB, XbB, S);

  // layer 2
  gemm_bf16_kernel<<<gD, gblk, 0, stream>>>(XbB, wqk2b, nullptr, nullptr, Qbuf, D_MODEL, sqd);
  gemm_bf16_kernel<<<gD, gblk, 0, stream>>>(XbB, wov2b, nullptr, nullptr, Vbuf, D_MODEL, 1.0f);
  attn_kernel<<<agrid, ablk, 0, stream>>>(Qbuf, XbB, Vbuf, XfB, XfA, XbA, S);

  // FFN residual: XfB = XfA + XbA @ WF^T (also bf16 copy in XbB)
  gemm_bf16_kernel<<<gD, gblk, 0, stream>>>(XbA, wfb, XfA, XfB, XbB, D_MODEL, 1.0f);

  // logits = XbB @ U^T -> d_out (fp32)
  dim3 gU(V / 64, M / 16);
  gemm_bf16_kernel<<<gU, gblk, 0, stream>>>(XbB, ub, nullptr, (float*)d_out, nullptr, V, 1.0f);
}
